// EGNNLayer_80444737454134
// MI455X (gfx1250) — matrix operand order repaired and checked by reading, not compiled, not run on hardware
//
#include <hip/hip_runtime.h>
#include <hip/hip_bf16.h>

// ---------------------------------------------------------------------------
// EGNN layer for MI455X (gfx1250): bf16 WMMA GEMMs (f32 accumulate), LDS-staged
// tiles, persistent workgroups (weights staged once per block), prefetch of the
// streamed edge data, f32 global atomics for the segment sums.
// ---------------------------------------------------------------------------

typedef __attribute__((ext_vector_type(16))) __bf16 v16bf;
typedef __attribute__((ext_vector_type(8)))  __bf16 v8bf;
typedef __attribute__((ext_vector_type(8)))  float  v8f;

#define NTHREADS 128   // 4 waves (wave32)
#define TILE_E   64    // edges per tile
#define TILE_N   64    // nodes per tile
#define HID      64
#define EIN_PAD  160   // 145 padded to 5 * 32

#define EDGE_BLOCKS 1024
#define NODE_BLOCKS 512

__device__ inline float silu_f(float v) { return v / (1.0f + __expf(-v)); }

// Stage a row-major f32 weight [K x 64] into LDS as bf16 in B-fragment-major
// order: frag index ((kc32*64 + n)*2 + hi)*16 + i  holds  W[kc32*32 + hi*16 + i][n].
// Rows K..Kpad-1 are zero-filled.
__device__ inline void stage_weight_frag(__bf16* dstf, const float* __restrict__ W,
                                         int K, int Kpad, int tid) {
  for (int idx = tid; idx < Kpad * 64; idx += NTHREADS) {
    int k = idx >> 6, n = idx & 63;
    float v = (k < K) ? W[k * 64 + n] : 0.0f;
    int kc = k >> 5, hi = (k >> 4) & 1, i = k & 15;
    dstf[((kc * 64 + n) * 2 + hi) * 16 + i] = (__bf16)v;
  }
}

// A-fragment (16-bit A 16x32, contiguous K to match B): lanes 0-15 = rows M, K base kc;
// lanes 16-31 = same rows, K base kc+16. Elements 0-15 -> K+0..15.
__device__ inline v16bf load_a_frag(const __bf16* Abase, int stride, int m0,
                                    int kc, int lane) {
  int row = m0 + (lane & 15);
  int kb  = kc + ((lane >> 4) << 4);
  const __bf16* p = Abase + row * stride + kb;
  v8bf lo = *(const v8bf*)p;
  v8bf hh = *(const v8bf*)(p + 8);
  v16bf a;
#pragma unroll
  for (int i = 0; i < 8; ++i) { a[i] = lo[i]; a[8 + i] = hh[i]; }
  return a;
}

// B-fragment: lane n (0-15) holds K = kc..kc+15, lanes 16-31 hold K = kc+16..kc+31
// (contiguous 32B per lane thanks to fragment-major staging).
__device__ inline v16bf load_b_frag(const __bf16* Wf, int kc32, int n0, int lane) {
  int n  = n0 + (lane & 15);
  int hi = lane >> 4;
  return *(const v16bf*)&Wf[((kc32 * 64 + n) * 2 + hi) * 16];
}

__device__ inline v8f wmma_bf16(v16bf a, v16bf b, v8f c) {
  return __builtin_amdgcn_wmma_f32_16x16x32_bf16(false, a, false, b,
                                                 (short)0, c, false, false);
}

// ---------------------------------------------------------------------------
__global__ void zero_ws_kernel(float* __restrict__ p, long long n) {
  long long i = (long long)blockIdx.x * blockDim.x + threadIdx.x;
  if (i < n) p[i] = 0.0f;
}

// ---------------------------------------------------------------------------
// Edge kernel (persistent): per tile of 64 edges
//   m = silu(silu([x_d|x_s|r2|ea] @ We1 + b1) @ We2 + b2)
//   msum[dst] += m ; deg[dst] += 1 ; coord[dst] += (m@Wc + bc) * dir
// ---------------------------------------------------------------------------
__global__ __launch_bounds__(NTHREADS) void egnn_edge_kernel(
    const float* __restrict__ x, const float* __restrict__ pos,
    const float* __restrict__ eattr, const long long* __restrict__ eidx,
    const float* __restrict__ We1, const float* __restrict__ be1,
    const float* __restrict__ We2, const float* __restrict__ be2,
    const float* __restrict__ Wc, const float* __restrict__ bc,
    float* __restrict__ msum, float* __restrict__ deg,
    float* __restrict__ coord, int nE) {
  __shared__ __align__(32) __bf16 We1f[EIN_PAD * 64];      // 20 KB
  __shared__ __align__(32) __bf16 We2f[64 * 64];           //  8 KB
  __shared__ __align__(32) __bf16 Abuf[TILE_E * EIN_PAD];  // 20 KB
  __shared__ __align__(32) __bf16 Mb[TILE_E * 64];         //  8 KB
  __shared__ float Mf[TILE_E * 64];                        // 16 KB
  __shared__ float b1s[64], b2s[64], Wcs[64];
  __shared__ float r2s[TILE_E], dirx[TILE_E], diry[TILE_E], dirz[TILE_E];
  __shared__ int srcs[TILE_E], dsts[TILE_E];

  const int tid  = threadIdx.x;
  const int lane = tid & 31;
  const int wave = tid >> 5;
  const int m0   = wave * 16;
  const int hi8  = (lane >> 4) << 3;
  const int ncl  = lane & 15;

  // --- Stage weights once per persistent block -----------------------------
  stage_weight_frag(We1f, We1, 145, EIN_PAD, tid);
  stage_weight_frag(We2f, We2, 64, 64, tid);
  if (tid < 64) { b1s[tid] = be1[tid]; b2s[tid] = be2[tid]; Wcs[tid] = Wc[tid]; }

  const int nTiles = (nE + TILE_E - 1) / TILE_E;

  for (int tile = blockIdx.x; tile < nTiles; tile += gridDim.x) {
    const int e0 = tile * TILE_E;

    // Prefetch next tile's streamed edge data (global_prefetch_b8).
    {
      int tnext = tile + gridDim.x;
      if (tnext < nTiles && tid < TILE_E) {
        int en = tnext * TILE_E + tid;
        __builtin_prefetch(&eattr[(long long)en * 16], 0, 1);
        __builtin_prefetch(&eidx[en], 0, 1);
        __builtin_prefetch(&eidx[(long long)nE + en], 0, 1);
      }
    }

    // --- Geometry + indices ------------------------------------------------
    if (tid < TILE_E) {
      int ec = e0 + tid; if (ec >= nE) ec = nE - 1;
      long long s = eidx[ec];
      long long d = eidx[(long long)nE + ec];
      srcs[tid] = (int)s; dsts[tid] = (int)d;
      float dx = pos[d * 3 + 0] - pos[s * 3 + 0];
      float dy = pos[d * 3 + 1] - pos[s * 3 + 1];
      float dz = pos[d * 3 + 2] - pos[s * 3 + 2];
      float r2 = dx * dx + dy * dy + dz * dz;
      float ri = rsqrtf(r2 + 1e-8f);
      r2s[tid] = r2; dirx[tid] = dx * ri; diry[tid] = dy * ri; dirz[tid] = dz * ri;
    }
    __syncthreads();

    // --- Build bf16 A tile [64 x 160] (x gathers hit L2: x is 12.8 MB) -----
    for (int idx = tid; idx < TILE_E * EIN_PAD; idx += NTHREADS) {
      int e = idx / EIN_PAD, c = idx - e * EIN_PAD;
      float v;
      if (c < 64)        v = x[(long long)dsts[e] * 64 + c];
      else if (c < 128)  v = x[(long long)srcs[e] * 64 + (c - 64)];
      else if (c == 128) v = r2s[e];
      else if (c < 145) {
        int ec = e0 + e; if (ec >= nE) ec = nE - 1;
        v = eattr[(long long)ec * 16 + (c - 129)];
      } else v = 0.0f;
      Abuf[idx] = (__bf16)v;
    }
    __syncthreads();

    // --- GEMM1: [64x160] x [160x64] -> silu -> Mb (bf16) -------------------
    v8f acc[4];
#pragma unroll
    for (int t = 0; t < 4; ++t)
#pragma unroll
      for (int r = 0; r < 8; ++r) acc[t][r] = 0.0f;
#pragma unroll
    for (int kc = 0; kc < EIN_PAD; kc += 32) {
      v16bf a = load_a_frag(Abuf, EIN_PAD, m0, kc, lane);
#pragma unroll
      for (int t = 0; t < 4; ++t) {
        v16bf b = load_b_frag(We1f, kc >> 5, t * 16, lane);
        acc[t] = wmma_bf16(a, b, acc[t]);
      }
    }
#pragma unroll
    for (int t = 0; t < 4; ++t) {
      float bb = b1s[t * 16 + ncl];
#pragma unroll
      for (int r = 0; r < 8; ++r) {
        float v = silu_f(acc[t][r] + bb);
        Mb[(m0 + r + hi8) * 64 + (t * 16 + ncl)] = (__bf16)v;
      }
    }
    __syncthreads();

    // --- GEMM2: [64x64] x [64x64] -> silu -> Mf (f32) ----------------------
#pragma unroll
    for (int t = 0; t < 4; ++t)
#pragma unroll
      for (int r = 0; r < 8; ++r) acc[t][r] = 0.0f;
#pragma unroll
    for (int kc = 0; kc < 64; kc += 32) {
      v16bf a = load_a_frag(Mb, 64, m0, kc, lane);
#pragma unroll
      for (int t = 0; t < 4; ++t) {
        v16bf b = load_b_frag(We2f, kc >> 5, t * 16, lane);
        acc[t] = wmma_bf16(a, b, acc[t]);
      }
    }
#pragma unroll
    for (int t = 0; t < 4; ++t) {
      float bb = b2s[t * 16 + ncl];
#pragma unroll
      for (int r = 0; r < 8; ++r) {
        float v = silu_f(acc[t][r] + bb);
        Mf[(m0 + r + hi8) * 64 + (t * 16 + ncl)] = v;
      }
    }
    __syncthreads();

    // --- Scatter m into msum (f32 atomics, L2-resident accumulators) -------
    for (int idx = tid; idx < TILE_E * 64; idx += NTHREADS) {
      int e = idx >> 6, n = idx & 63;
      if (e0 + e < nE)
        atomicAdd(&msum[(long long)dsts[e] * 64 + n], Mf[idx]);
    }
    // gamma = m @ Wc + bc ; degree and coordinate scatter
    if (tid < TILE_E && (e0 + tid) < nE) {
      float g = bc[0];
#pragma unroll 8
      for (int k = 0; k < 64; ++k) g += Mf[tid * 64 + k] * Wcs[k];
      int d = dsts[tid];
      atomicAdd(&deg[d], 1.0f);
      atomicAdd(&coord[(long long)d * 3 + 0], g * dirx[tid]);
      atomicAdd(&coord[(long long)d * 3 + 1], g * diry[tid]);
      atomicAdd(&coord[(long long)d * 3 + 2], g * dirz[tid]);
    }
    __syncthreads();  // srcs/dsts/dirs/Mf are rewritten next iteration
  }
}

// ---------------------------------------------------------------------------
// Node kernel (persistent): h = silu([x | msum/deg] @ Wn1 + b1);
// x_out = h @ Wn2 + b2; pos_out = pos + coord/deg
// ---------------------------------------------------------------------------
__global__ __launch_bounds__(NTHREADS) void egnn_node_kernel(
    const float* __restrict__ x, const float* __restrict__ pos,
    const float* __restrict__ Wn1, const float* __restrict__ bn1,
    const float* __restrict__ Wn2, const float* __restrict__ bn2,
    const float* __restrict__ msum, const float* __restrict__ deg,
    const float* __restrict__ coord, float* __restrict__ out, int nN) {
  __shared__ __align__(32) __bf16 Wn1f[128 * 64];        // 16 KB
  __shared__ __align__(32) __bf16 Wn2f[64 * 64];         //  8 KB
  __shared__ __align__(32) __bf16 Abuf[TILE_N * 128];    // 16 KB
  __shared__ __align__(32) __bf16 Hb[TILE_N * 64];       //  8 KB
  __shared__ float b1s[64], b2s[64], invdeg[TILE_N];

  const int tid  = threadIdx.x;
  const int lane = tid & 31;
  const int wave = tid >> 5;
  const int m0   = wave * 16;
  const int hi8  = (lane >> 4) << 3;
  const int ncl  = lane & 15;

  stage_weight_frag(Wn1f, Wn1, 128, 128, tid);
  stage_weight_frag(Wn2f, Wn2, 64, 64, tid);
  if (tid < 64) { b1s[tid] = bn1[tid]; b2s[tid] = bn2[tid]; }

  const int nTiles = (nN + TILE_N - 1) / TILE_N;

  for (int tile = blockIdx.x; tile < nTiles; tile += gridDim.x) {
    const int base = tile * TILE_N;

    if (tid < TILE_N) {
      int nd = base + tid; if (nd >= nN) nd = nN - 1;
      invdeg[tid] = 1.0f / fmaxf(deg[nd], 1.0f);
    }
    __syncthreads();

    // A tile [64 x 128] = [x | msum/deg]
    for (int idx = tid; idx < TILE_N * 128; idx += NTHREADS) {
      int e = idx >> 7, c = idx & 127;
      int nd = base + e; if (nd >= nN) nd = nN - 1;
      float v = (c < 64) ? x[(long long)nd * 64 + c]
                         : msum[(long long)nd * 64 + (c - 64)] * invdeg[e];
      Abuf[idx] = (__bf16)v;
    }
    __syncthreads();

    v8f acc[4];
#pragma unroll
    for (int t = 0; t < 4; ++t)
#pragma unroll
      for (int r = 0; r < 8; ++r) acc[t][r] = 0.0f;
#pragma unroll
    for (int kc = 0; kc < 128; kc += 32) {
      v16bf a = load_a_frag(Abuf, 128, m0, kc, lane);
#pragma unroll
      for (int t = 0; t < 4; ++t) {
        v16bf b = load_b_frag(Wn1f, kc >> 5, t * 16, lane);
        acc[t] = wmma_bf16(a, b, acc[t]);
      }
    }
#pragma unroll
    for (int t = 0; t < 4; ++t) {
      float bb = b1s[t * 16 + ncl];
#pragma unroll
      for (int r = 0; r < 8; ++r) {
        float v = silu_f(acc[t][r] + bb);
        Hb[(m0 + r + hi8) * 64 + (t * 16 + ncl)] = (__bf16)v;
      }
    }
    __syncthreads();

#pragma unroll
    for (int t = 0; t < 4; ++t)
#pragma unroll
      for (int r = 0; r < 8; ++r) acc[t][r] = 0.0f;
#pragma unroll
    for (int kc = 0; kc < 64; kc += 32) {
      v16bf a = load_a_frag(Hb, 64, m0, kc, lane);
#pragma unroll
      for (int t = 0; t < 4; ++t) {
        v16bf b = load_b_frag(Wn2f, kc >> 5, t * 16, lane);
        acc[t] = wmma_bf16(a, b, acc[t]);
      }
    }
    // x_out writeback (bounds-guarded; divergence is fine here, WMMAs done)
#pragma unroll
    for (int t = 0; t < 4; ++t) {
      float bb = b2s[t * 16 + ncl];
#pragma unroll
      for (int r = 0; r < 8; ++r) {
        int node = base + m0 + r + hi8;
        if (node < nN)
          out[(long long)node * 64 + (t * 16 + ncl)] = acc[t][r] + bb;
      }
    }
    // pos_out = pos + coord/deg   (after x_out region: offset nN*64)
    if (tid < TILE_N) {
      int node = base + tid;
      if (node < nN) {
        float iv = invdeg[tid];
        long long po = (long long)nN * 64 + (long long)node * 3;
#pragma unroll
        for (int k = 0; k < 3; ++k)
          out[po + k] = pos[(long long)node * 3 + k] +
                        coord[(long long)node * 3 + k] * iv;
      }
    }
    __syncthreads();  // invdeg/Abuf/Hb rewritten next iteration
  }
}

// ---------------------------------------------------------------------------
extern "C" void kernel_launch(void* const* d_in, const int* in_sizes, int n_in,
                              void* d_out, int out_size, void* d_ws, size_t ws_size,
                              hipStream_t stream) {
  const float*     x     = (const float*)d_in[0];
  const float*     pos   = (const float*)d_in[1];
  const float*     eattr = (const float*)d_in[2];
  const long long* eidx  = (const long long*)d_in[3];
  const float*     We1   = (const float*)d_in[4];
  const float*     be1   = (const float*)d_in[5];
  const float*     We2   = (const float*)d_in[6];
  const float*     be2   = (const float*)d_in[7];
  const float*     Wn1   = (const float*)d_in[8];
  const float*     bn1   = (const float*)d_in[9];
  const float*     Wn2   = (const float*)d_in[10];
  const float*     bn2   = (const float*)d_in[11];
  const float*     Wc    = (const float*)d_in[12];
  const float*     bc    = (const float*)d_in[13];

  const int N = in_sizes[0] / 64;   // 50000
  const int E = in_sizes[3] / 2;    // 1000000

  // workspace: msum[N*64] | deg[N] | coord[N*3]  (contiguous, ~13.6 MB)
  float* msum  = (float*)d_ws;
  float* deg   = msum + (size_t)N * 64;
  float* coord = deg + N;

  long long zn = (long long)N * 68;
  zero_ws_kernel<<<(int)((zn + 255) / 256), 256, 0, stream>>>(msum, zn);

  int eTiles = (E + TILE_E - 1) / TILE_E;
  int eGrid  = eTiles < EDGE_BLOCKS ? eTiles : EDGE_BLOCKS;
  egnn_edge_kernel<<<eGrid, NTHREADS, 0, stream>>>(
      x, pos, eattr, eidx, We1, be1, We2, be2, Wc, bc, msum, deg, coord, E);

  int nTiles = (N + TILE_N - 1) / TILE_N;
  int nGrid  = nTiles < NODE_BLOCKS ? nTiles : NODE_BLOCKS;
  egnn_node_kernel<<<nGrid, NTHREADS, 0, stream>>>(
      x, pos, Wn1, bn1, Wn2, bn2, msum, deg, coord, (float*)d_out, N);
}
